// WindowAttention_77644418777280
// MI455X (gfx1250) — matrix operand order repaired and checked by reading, not compiled, not run on hardware
//
#include <hip/hip_runtime.h>
#include <hip/hip_bf16.h>

// MI455X (gfx1250) wave32 WMMA implementation of Swin-V2 window attention.
// Heavy GEMMs (QKV proj, q@k^T, P@V, output proj) on v_wmma_f32_16x16x32_bf16,
// fp32 accumulation, 2x2 register blocking, activation tiles staged to LDS via
// the Tensor Data Mover (tensor_load_to_lds + s_wait_tensorcnt).
// Workspace requirement: ~540 MB (xbf region reused for attention output).

typedef __bf16 bf16;
typedef __attribute__((ext_vector_type(16))) __bf16 v16bf;
typedef __attribute__((ext_vector_type(8)))  float  v8f;
typedef __attribute__((ext_vector_type(4)))  unsigned int v4u;
typedef __attribute__((ext_vector_type(8)))  int v8i;
typedef __attribute__((ext_vector_type(4)))  int v4i;

#define BW    2048
#define NTOK  49
#define NPAD  64
#define CDIM  512
#define NHEAD 16
#define HDIM  32
#define NWIN  64
#define QKVN  (3 * CDIM)

#if __has_builtin(__builtin_amdgcn_tensor_load_to_lds) && __has_builtin(__builtin_amdgcn_s_wait_tensorcnt)
#define HAVE_TDM 1
#else
#define HAVE_TDM 0
#endif

// ---------- WMMA fragment helpers (layouts per CDNA5 ISA) ----------

// A matrix 16x32 bf16: lane = 16*half + M; per-lane K = 8*half+{0..7} and 16+8*half+{0..7}
static __device__ __forceinline__ v16bf load_fragA(const bf16* __restrict__ base, int ld, int lane) {
  const int m = lane & 15, half = lane >> 4;
  const bf16* p = base + m * ld + half * 8;
  v16bf f;
#pragma unroll
  for (int j = 0; j < 8; ++j) f[j] = p[j];
#pragma unroll
  for (int j = 0; j < 8; ++j) f[8 + j] = p[16 + j];
  return f;
}

// B matrix 32x16 bf16 (KxN), stored transposed: row n holds K contiguous.
// per-lane K = 8*half+{0..7} and 16+8*half+{0..7} (matches fragA)
static __device__ __forceinline__ v16bf load_fragB(const bf16* __restrict__ baseT, int ld, int lane) {
  const int n = lane & 15, half = lane >> 4;
  const bf16* p = baseT + n * ld + half * 8;
  v16bf f;
#pragma unroll
  for (int j = 0; j < 8; ++j) f[j] = p[j];
#pragma unroll
  for (int j = 0; j < 8; ++j) f[8 + j] = p[16 + j];
  return f;
}

static __device__ __forceinline__ v8f wmma_bf16(v16bf a, v16bf b, v8f c) {
  return __builtin_amdgcn_wmma_f32_16x16x32_bf16(false, a, false, b, (short)0, c, false, false);
}

// ---------- TDM: 2D bf16 tile Global -> LDS (D# per CDNA5 ISA) ----------

#if HAVE_TDM
static __device__ __forceinline__ void tdm_load_2d_bf16(unsigned int lds_off, const void* gptr,
                                                        unsigned int dim0, unsigned int dim1,
                                                        unsigned int stride0) {
  const unsigned long long ga = (unsigned long long)(uintptr_t)gptr;
  v4u g0;
  g0[0] = 1u;                                           // count=1, user mode, no gather
  g0[1] = lds_off;                                      // lds_addr (bytes)
  g0[2] = (unsigned int)(ga & 0xffffffffu);             // global_addr[31:0]
  g0[3] = (unsigned int)((ga >> 32) & 0x01ffffffu) | (2u << 30);  // addr[56:32], type=2
  v8i g1;
  g1[0] = (int)(1u << 16);                              // wg_mask=0, data_size=1 (2 bytes)
  g1[1] = (int)((dim0 & 0xffffu) << 16);                // tensor_dim0[15:0]
  g1[2] = (int)(((dim0 >> 16) & 0xffffu) | ((dim1 & 0xffffu) << 16));  // td0 hi | td1 lo
  g1[3] = (int)(((dim1 >> 16) & 0xffffu) | ((dim0 & 0xffffu) << 16));  // td1 hi | tile_dim0
  g1[4] = (int)(dim1 & 0xffffu);                        // tile_dim1 | tile_dim2=0
  g1[5] = (int)stride0;                                 // tensor_dim0_stride[31:0]
  g1[6] = 0;                                            // stride0[47:32] | stride1 lo
  g1[7] = 0;
  v4i z4 = {0, 0, 0, 0};
  v8i z8 = {0, 0, 0, 0, 0, 0, 0, 0};
  // amdgpu-toolchain (clang-23) 6-arg form: (g0, g1, g2, g3, g4, cpol)
  __builtin_amdgcn_tensor_load_to_lds(g0, g1, z4, z4, z8, 0);
}
#endif

// Stage one [NPAD][CDIM] bf16 window into LDS; barrier included.
static __device__ __forceinline__ void stage_window_lds(bf16* dst, const bf16* src, int tid) {
#if HAVE_TDM
  if ((tid >> 5) == 0) {  // one wave issues the DMA; TDM ignores EXEC (redundant issue benign)
    tdm_load_2d_bf16((unsigned int)(uintptr_t)dst, src, CDIM, NPAD, CDIM);
    __builtin_amdgcn_s_wait_tensorcnt(0);
  }
#else
  for (int i = tid * 8; i < NPAD * CDIM; i += 256 * 8)
    *(uint4*)(dst + i) = *(const uint4*)(src + i);
#endif
  __syncthreads();
}

// ---------- Stage 0: conversions ----------

__global__ void k_convert_x(const float* __restrict__ x, bf16* __restrict__ xbf) {
  size_t idx = (size_t)blockIdx.x * blockDim.x + threadIdx.x;
  if (idx >= (size_t)BW * NPAD * CDIM) return;
  int c = (int)(idx % CDIM);
  int t = (int)((idx / CDIM) % NPAD);
  int b = (int)(idx / ((size_t)CDIM * NPAD));
  float v = (t < NTOK) ? x[((size_t)b * NTOK + t) * CDIM + c] : 0.0f;
  xbf[idx] = (bf16)v;
}

// W [K][N] fp32 -> WT [N][K] bf16
__global__ void k_transpose_w(const float* __restrict__ W, bf16* __restrict__ WT, int K, int N) {
  size_t idx = (size_t)blockIdx.x * blockDim.x + threadIdx.x;
  if (idx >= (size_t)K * N) return;
  int kk = (int)(idx % K);
  int n  = (int)(idx / K);
  WT[idx] = (bf16)W[(size_t)kk * N + n];
}

// ---------- Stage 1: RPE bias table -> pb[16][64][64] = 16*sigmoid(mlp) ----------

static __device__ __forceinline__ float normcoord(int v) {
  // (v / 11) * (8 / 11) then sign*log1p(|.|)/log(8)
  float x = (float)v * (8.0f / 121.0f);
  return copysignf(log1pf(fabsf(x)), x) * (1.0f / 2.0794415416798357f);
}

__global__ void k_rpe(const float* __restrict__ w1, const float* __restrict__ b1,
                      const float* __restrict__ w2, float* __restrict__ pb) {
  __shared__ float bt[169][NHEAD];
  for (int o = threadIdx.x; o < 169 * NHEAD; o += blockDim.x) {
    int p = o >> 4, h = o & 15;
    int i = p / 13, j = p % 13;
    float c0 = normcoord(j - 6);  // table[...,0] = ch[j]
    float c1 = normcoord(i - 6);  // table[...,1] = ch[i]
    float s = 0.0f;
    for (int r = 0; r < CDIM; ++r) {
      float hv = c0 * w1[r] + c1 * w1[CDIM + r] + b1[r];
      hv = fmaxf(hv, 0.0f);
      s += hv * w2[r * NHEAD + h];
    }
    bt[p][h] = s;
  }
  __syncthreads();
  for (int o = threadIdx.x; o < NHEAD * NPAD * NPAD; o += blockDim.x) {
    int h = o >> 12, m = (o >> 6) & 63, n = o & 63;
    float v = 0.0f;
    if (m < NTOK && n < NTOK) {
      int d0 = (m % 7) - (n % 7);
      int d1 = (m / 7) - (n / 7);
      int idx = (d0 + 6) * 13 + (d1 + 6);
      v = 16.0f / (1.0f + __expf(-bt[idx][h]));
    }
    pb[o] = v;
  }
}

// ---------- Stage 2: QKV projection GEMM per window (2x2 register blocking) ----------

static __device__ __forceinline__ void store_qkv_tile(
    const v8f& acc, int b, int tm, int tn, int lane,
    const float* __restrict__ qb, const float* __restrict__ vb,
    bf16* __restrict__ qws, bf16* __restrict__ kws, bf16* __restrict__ vtws) {
  const int n = lane & 15, half = lane >> 4;
  const int fin = (tn * 16 + n) & 511;
  const int sec = __builtin_amdgcn_readfirstlane(tn >> 5);  // uniform: 0=q,1=k,2=v
  const int h = fin >> 5, d = fin & 31;
  const size_t hb = (size_t)b * NHEAD + h;
  if (sec == 0) {
    const float bias = qb[fin];
#pragma unroll
    for (int r = 0; r < 8; ++r) {
      const int m = tm * 16 + r + 8 * half;
      qws[(hb * NPAD + m) * HDIM + d] = (bf16)(acc[r] + bias);
    }
  } else if (sec == 1) {
#pragma unroll
    for (int r = 0; r < 8; ++r) {
      const int m = tm * 16 + r + 8 * half;
      kws[(hb * NPAD + m) * HDIM + d] = (bf16)acc[r];
    }
  } else {
    const float bias = vb[fin];
#pragma unroll
    for (int r = 0; r < 8; ++r) {
      const int m = tm * 16 + r + 8 * half;
      vtws[(hb * HDIM + d) * NPAD + m] = (bf16)(acc[r] + bias);
    }
  }
}

__global__ __launch_bounds__(256) void k_qkv(
    const bf16* __restrict__ xbf,   // [BW][64][512]
    const bf16* __restrict__ wT,    // [1536][512]
    const float* __restrict__ qb, const float* __restrict__ vb,
    bf16* __restrict__ qws,         // [BW][16][64][32]
    bf16* __restrict__ kws,         // [BW][16][64][32]
    bf16* __restrict__ vtws)        // [BW][16][32][64]
{
  extern __shared__ char smem[];
  bf16* xs = (bf16*)smem;           // [64][512]
  const int b = blockIdx.x;
  const int wave = threadIdx.x >> 5;
  const int lane = threadIdx.x & 31;
  stage_window_lds(xs, xbf + (size_t)b * NPAD * CDIM, threadIdx.x);

  // 2 (M-pairs) x 48 (N-pairs) blocks of 2x2 16x16 tiles
  for (int blk = wave; blk < 96; blk += 8) {
    const int tm0 = (blk & 1) * 2;
    const int tn0 = (blk >> 1) * 2;
    const bf16* bT0 = wT + (size_t)(tn0 * 16) * CDIM;
    const bf16* bT1 = bT0 + (size_t)16 * CDIM;
    __builtin_prefetch(bT0, 0, 0);
    __builtin_prefetch(bT1, 0, 0);
    v8f a00 = {}, a01 = {}, a10 = {}, a11 = {};
#pragma unroll 4
    for (int kk = 0; kk < 16; ++kk) {
      v16bf fa0 = load_fragA(xs + (size_t)(tm0 * 16) * CDIM + kk * 32, CDIM, lane);
      v16bf fa1 = load_fragA(xs + (size_t)(tm0 * 16 + 16) * CDIM + kk * 32, CDIM, lane);
      v16bf fb0 = load_fragB(bT0 + kk * 32, CDIM, lane);
      v16bf fb1 = load_fragB(bT1 + kk * 32, CDIM, lane);
      a00 = wmma_bf16(fa0, fb0, a00);
      a01 = wmma_bf16(fa0, fb1, a01);
      a10 = wmma_bf16(fa1, fb0, a10);
      a11 = wmma_bf16(fa1, fb1, a11);
    }
    store_qkv_tile(a00, b, tm0,     tn0,     lane, qb, vb, qws, kws, vtws);
    store_qkv_tile(a01, b, tm0,     tn0 + 1, lane, qb, vb, qws, kws, vtws);
    store_qkv_tile(a10, b, tm0 + 1, tn0,     lane, qb, vb, qws, kws, vtws);
    store_qkv_tile(a11, b, tm0 + 1, tn0 + 1, lane, qb, vb, qws, kws, vtws);
  }
}

// ---------- Stage 3: L2-normalize q,k rows; fold logit scale into q ----------

__global__ void k_norm(bf16* __restrict__ q, bf16* __restrict__ k,
                       const float* __restrict__ logit_scale) {
  size_t row = (size_t)blockIdx.x * blockDim.x + threadIdx.x;
  if (row >= (size_t)BW * NHEAD * NPAD) return;
  const int h = (int)((row >> 6) & 15);
  const float sc = __expf(fminf(logit_scale[h], 4.60517019f));  // ln(100)
  float tmp[HDIM];
  bf16* qr = q + row * HDIM;
  float ss = 0.0f;
#pragma unroll
  for (int i = 0; i < HDIM; ++i) { tmp[i] = (float)qr[i]; ss += tmp[i] * tmp[i]; }
  float r = rsqrtf(fmaxf(ss, 1e-12f)) * sc;
#pragma unroll
  for (int i = 0; i < HDIM; ++i) qr[i] = (bf16)(tmp[i] * r);

  bf16* kr = k + row * HDIM;
  ss = 0.0f;
#pragma unroll
  for (int i = 0; i < HDIM; ++i) { tmp[i] = (float)kr[i]; ss += tmp[i] * tmp[i]; }
  r = rsqrtf(fmaxf(ss, 1e-12f));
#pragma unroll
  for (int i = 0; i < HDIM; ++i) kr[i] = (bf16)(tmp[i] * r);
}

// ---------- Stage 4: attention per (window, head) ----------

__global__ __launch_bounds__(128) void k_attn(
    const bf16* __restrict__ qws, const bf16* __restrict__ kws,
    const bf16* __restrict__ vtws, const float* __restrict__ pb,   // [16][64][64]
    const float* __restrict__ mask,                                 // [64][49][49]
    bf16* __restrict__ aout)                                        // [BW][64][512]
{
  __shared__ float S[NPAD][NPAD];
  __shared__ bf16  P[NPAD][NPAD];
  const int b = blockIdx.x >> 4;
  const int h = blockIdx.x & 15;
  const int wave = threadIdx.x >> 5;   // 0..3 -> M tile
  const int lane = threadIdx.x & 31;
  const bf16* qh = qws  + ((size_t)b * NHEAD + h) * (NPAD * HDIM);  // [64][32]
  const bf16* kh = kws  + ((size_t)b * NHEAD + h) * (NPAD * HDIM);  // [64][32]
  const bf16* vh = vtws + ((size_t)b * NHEAD + h) * (HDIM * NPAD);  // [32][64]
  const float* pbh = pb + h * NPAD * NPAD;
  const float* mk  = mask + (size_t)(b & (NWIN - 1)) * NTOK * NTOK;

  const int tm = wave;
  const int n = lane & 15, half = lane >> 4;
  v16bf a = load_fragA(qh + (size_t)(tm * 16) * HDIM, HDIM, lane);
#pragma unroll
  for (int tn = 0; tn < 4; ++tn) {
    v16bf bb = load_fragB(kh + (size_t)(tn * 16) * HDIM, HDIM, lane);
    v8f acc = {};
    acc = wmma_bf16(a, bb, acc);
#pragma unroll
    for (int r = 0; r < 8; ++r) {
      const int m = tm * 16 + r + 8 * half;
      const int nn = tn * 16 + n;
      float s = acc[r] + pbh[m * NPAD + nn];
      if (m < NTOK && nn < NTOK) s += mk[m * NTOK + nn];
      S[m][nn] = s;
    }
  }
  __syncthreads();
  // row softmax over first 49 columns
  if (threadIdx.x < NPAD) {
    const int row = threadIdx.x;
    if (row < NTOK) {
      float mx = -3.4e38f;
      for (int c = 0; c < NTOK; ++c) mx = fmaxf(mx, S[row][c]);
      float sum = 0.0f;
      for (int c = 0; c < NTOK; ++c) sum += __expf(S[row][c] - mx);
      const float inv = 1.0f / sum;
      for (int c = 0; c < NPAD; ++c)
        P[row][c] = (c < NTOK) ? (bf16)(__expf(S[row][c] - mx) * inv) : (bf16)0.0f;
    } else {
      for (int c = 0; c < NPAD; ++c) P[row][c] = (bf16)0.0f;
    }
  }
  __syncthreads();
  // O = P[64x64] @ V[64x32]
#pragma unroll
  for (int tn = 0; tn < 2; ++tn) {
    v8f acc = {};
#pragma unroll
    for (int ks = 0; ks < 2; ++ks) {
      v16bf pa = load_fragA(&P[tm * 16][ks * 32], NPAD, lane);
      v16bf bb = load_fragB(vh + (size_t)(tn * 16) * NPAD + ks * 32, NPAD, lane);
      acc = wmma_bf16(pa, bb, acc);
    }
#pragma unroll
    for (int r = 0; r < 8; ++r) {
      const int m = tm * 16 + r + 8 * half;
      const int d = tn * 16 + n;
      aout[((size_t)b * NPAD + m) * CDIM + h * HDIM + d] = (bf16)acc[r];
    }
  }
}

// ---------- Stage 5: output projection GEMM per window (2x2 register blocking) ----------

__global__ __launch_bounds__(256) void k_proj(
    const bf16* __restrict__ aout,   // [BW][64][512]
    const bf16* __restrict__ pwT,    // [512][512]
    const float* __restrict__ pbias,
    float* __restrict__ out)         // [BW][49][512]
{
  extern __shared__ char smem[];
  bf16* as = (bf16*)smem;            // [64][512]
  const int b = blockIdx.x;
  const int wave = threadIdx.x >> 5;
  const int lane = threadIdx.x & 31;
  stage_window_lds(as, aout + (size_t)b * NPAD * CDIM, threadIdx.x);

  // 2 (M-pairs) x 16 (N-pairs) blocks of 2x2 16x16 tiles
  for (int blk = wave; blk < 32; blk += 8) {
    const int tm0 = (blk & 1) * 2;
    const int tn0 = (blk >> 1) * 2;
    const bf16* bT0 = pwT + (size_t)(tn0 * 16) * CDIM;
    const bf16* bT1 = bT0 + (size_t)16 * CDIM;
    __builtin_prefetch(bT0, 0, 0);
    __builtin_prefetch(bT1, 0, 0);
    v8f a00 = {}, a01 = {}, a10 = {}, a11 = {};
#pragma unroll 4
    for (int kk = 0; kk < 16; ++kk) {
      v16bf fa0 = load_fragA(as + (size_t)(tm0 * 16) * CDIM + kk * 32, CDIM, lane);
      v16bf fa1 = load_fragA(as + (size_t)(tm0 * 16 + 16) * CDIM + kk * 32, CDIM, lane);
      v16bf fb0 = load_fragB(bT0 + kk * 32, CDIM, lane);
      v16bf fb1 = load_fragB(bT1 + kk * 32, CDIM, lane);
      a00 = wmma_bf16(fa0, fb0, a00);
      a01 = wmma_bf16(fa0, fb1, a01);
      a10 = wmma_bf16(fa1, fb0, a10);
      a11 = wmma_bf16(fa1, fb1, a11);
    }
    const int n = lane & 15, half = lane >> 4;
#pragma unroll
    for (int j = 0; j < 4; ++j) {
      const v8f& acc = (j == 0) ? a00 : (j == 1) ? a01 : (j == 2) ? a10 : a11;
      const int tm = tm0 + (j >> 1);
      const int ng = (tn0 + (j & 1)) * 16 + n;
      const float bias = pbias[ng];
#pragma unroll
      for (int r = 0; r < 8; ++r) {
        const int m = tm * 16 + r + 8 * half;
        if (m < NTOK) out[((size_t)b * NTOK + m) * CDIM + ng] = acc[r] + bias;
      }
    }
  }
}

// ---------- host ----------

extern "C" void kernel_launch(void* const* d_in, const int* in_sizes, int n_in,
                              void* d_out, int out_size, void* d_ws, size_t ws_size,
                              hipStream_t stream) {
  const float* x     = (const float*)d_in[0];
  const float* mask  = (const float*)d_in[1];
  const float* qkvw  = (const float*)d_in[2];
  const float* qb    = (const float*)d_in[3];
  const float* vb    = (const float*)d_in[4];
  const float* lsc   = (const float*)d_in[5];
  const float* w1    = (const float*)d_in[6];
  const float* b1    = (const float*)d_in[7];
  const float* w2    = (const float*)d_in[8];
  const float* pw    = (const float*)d_in[9];
  const float* pbias = (const float*)d_in[10];
  float* out = (float*)d_out;

  char* base = (char*)d_ws;
  size_t off = 0;
  auto carve = [&](size_t bytes) -> void* {
    void* p = base + off;
    off += (bytes + 255) & ~(size_t)255;
    return p;
  };
  bf16* xbf    = (bf16*)carve((size_t)BW * NPAD * CDIM * sizeof(bf16));  // reused as aout
  bf16* qkvwT  = (bf16*)carve((size_t)QKVN * CDIM * sizeof(bf16));
  bf16* projwT = (bf16*)carve((size_t)CDIM * CDIM * sizeof(bf16));
  float* pbt   = (float*)carve((size_t)NHEAD * NPAD * NPAD * sizeof(float));
  bf16* qws    = (bf16*)carve((size_t)BW * NHEAD * NPAD * HDIM * sizeof(bf16));
  bf16* kws    = (bf16*)carve((size_t)BW * NHEAD * NPAD * HDIM * sizeof(bf16));
  bf16* vtws   = (bf16*)carve((size_t)BW * NHEAD * HDIM * NPAD * sizeof(bf16));
  bf16* aout   = xbf;  // xbf is dead after k_qkv; safe stream-ordered reuse

  {
    size_t n = (size_t)BW * NPAD * CDIM;
    k_convert_x<<<(unsigned)((n + 255) / 256), 256, 0, stream>>>(x, xbf);
  }
  k_transpose_w<<<(QKVN * CDIM + 255) / 256, 256, 0, stream>>>(qkvw, qkvwT, CDIM, QKVN);
  k_transpose_w<<<(CDIM * CDIM + 255) / 256, 256, 0, stream>>>(pw, projwT, CDIM, CDIM);
  k_rpe<<<1, 256, 0, stream>>>(w1, b1, w2, pbt);
  k_qkv<<<BW, 256, 64 * 1024, stream>>>(xbf, qkvwT, qb, vb, qws, kws, vtws);
  {
    size_t rows = (size_t)BW * NHEAD * NPAD;
    k_norm<<<(unsigned)((rows + 255) / 256), 256, 0, stream>>>(qws, kws, lsc);
  }
  k_attn<<<BW * NHEAD, 128, 0, stream>>>(qws, kws, vtws, pbt, mask, aout);
  k_proj<<<BW, 256, 64 * 1024, stream>>>(aout, projwT, pbias, out);
}
